// GraphAttentionLayer_3289944949193
// MI455X (gfx1250) — matrix operand order repaired and checked by reading, not compiled, not run on hardware
//
#include <hip/hip_runtime.h>

#define NN 4096
#define FF 512
#define GAT_ALPHA 0.2f
#define NEGINF (-1e12f)

typedef __attribute__((ext_vector_type(16))) __bf16 v16bf;
typedef __attribute__((ext_vector_type(8)))  float  v8f;
typedef __attribute__((ext_vector_type(8)))  unsigned short v8u;

union Frag { v16bf v; v8u h[2]; };

static __device__ __forceinline__ unsigned short f2bf(float f) {
  unsigned u = __float_as_uint(f);
  return (unsigned short)((u + 0x7FFFu + ((u >> 16) & 1u)) >> 16);
}

// ---------------------------------------------------------------------------
// Kernel 1: Wa1 = W @ a[:F], Wa2 = W @ a[F:]   (512 dot products of length 512)
// ---------------------------------------------------------------------------
__global__ void k_wa(const float* __restrict__ W, const float* __restrict__ a,
                     float* __restrict__ Wa1, float* __restrict__ Wa2) {
  int t = blockIdx.x * blockDim.x + threadIdx.x;
  if (t >= FF) return;
  const float* row = W + (long)t * FF;
  float a1 = 0.f, a2 = 0.f;
  for (int j = 0; j < FF; ++j) {
    float w = row[j];
    a1 += w * a[j];
    a2 += w * a[FF + j];
  }
  Wa1[t] = a1;
  Wa2[t] = a2;
}

// ---------------------------------------------------------------------------
// Kernel 2: s1[i] = inp[i,:] . Wa1 ; s2[i] = inp[i,:] . Wa2  (wave per row)
// ---------------------------------------------------------------------------
__global__ void k_scores(const float* __restrict__ inp, const float* __restrict__ Wa1,
                         const float* __restrict__ Wa2, float* __restrict__ s1,
                         float* __restrict__ s2) {
  const int lane = threadIdx.x & 31;
  const int wv   = threadIdx.x >> 5;
  const int row  = blockIdx.x * 8 + wv;
  const float* r = inp + (long)row * FF;
  float x1 = 0.f, x2 = 0.f;
  for (int j = lane; j < FF; j += 32) {
    float v = r[j];
    x1 += v * Wa1[j];
    x2 += v * Wa2[j];
  }
  for (int off = 16; off > 0; off >>= 1) {
    x1 += __shfl_down(x1, off, 32);
    x2 += __shfl_down(x2, off, 32);
  }
  if (lane == 0) { s1[row] = x1; s2[row] = x2; }
}

// ---------------------------------------------------------------------------
// Kernel 3a: inpT[f][j] = bf16(inp[j][f])   (WMMA-B friendly transposed bf16)
// ---------------------------------------------------------------------------
__global__ void k_cvt_inpT(const float* __restrict__ inp, unsigned short* __restrict__ inpT) {
  int idx = blockIdx.x * 256 + threadIdx.x;   // 512*4096 elements
  int f = idx >> 12, j = idx & (NN - 1);
  inpT[idx] = f2bf(inp[(long)j * FF + f]);
}

// Kernel 3b: wT[n][k] = bf16(wtrans[k][n])   (512 x 1024)
__global__ void k_cvt_wT(const float* __restrict__ wtrans, unsigned short* __restrict__ wT) {
  int idx = blockIdx.x * 256 + threadIdx.x;   // 512*1024 elements
  int n = idx >> 10, k = idx & 1023;
  wT[idx] = f2bf(wtrans[(long)k * FF + n]);
}

// ---------------------------------------------------------------------------
// Kernel 4: per-row online softmax stats for positive and negated logits.
// ---------------------------------------------------------------------------
__global__ void k_stats(const float* __restrict__ adj, const float* __restrict__ s1g,
                        const float* __restrict__ s2g, float* __restrict__ mp,
                        float* __restrict__ ilp, float* __restrict__ mn,
                        float* __restrict__ iln) {
  const int lane = threadIdx.x & 31;
  const int wv   = threadIdx.x >> 5;
  const int row  = blockIdx.x * 8 + wv;
  const float s1v = s1g[row];
  const float* arow = adj + (long)row * NN;
  float m1 = -3.0e38f, l1 = 0.f, m2 = -3.0e38f, l2 = 0.f;
  for (int j = lane; j < NN; j += 32) {
    float x = s1v + s2g[j];
    float e = x > 0.f ? x : GAT_ALPHA * x;
    float aj = arow[j];
    float fp = aj > 0.f ? e  : NEGINF;
    float fn = aj > 0.f ? -e : NEGINF;
    float nm = fmaxf(m1, fp); l1 = l1 * __expf(m1 - nm) + __expf(fp - nm); m1 = nm;
    nm = fmaxf(m2, fn);       l2 = l2 * __expf(m2 - nm) + __expf(fn - nm); m2 = nm;
  }
  for (int off = 16; off > 0; off >>= 1) {
    float om = __shfl_down(m1, off, 32), ol = __shfl_down(l1, off, 32);
    float nm = fmaxf(m1, om); l1 = l1 * __expf(m1 - nm) + ol * __expf(om - nm); m1 = nm;
    om = __shfl_down(m2, off, 32); ol = __shfl_down(l2, off, 32);
    nm = fmaxf(m2, om); l2 = l2 * __expf(m2 - nm) + ol * __expf(om - nm); m2 = nm;
  }
  if (lane == 0) { mp[row] = m1; ilp[row] = 1.f / l1; mn[row] = m2; iln[row] = 1.f / l2; }
}

// ---------------------------------------------------------------------------
// Kernel 5: fused attention-probability + GEMM.
// Block = 512 threads (16 waves) = 32-row M-tile x full N=512.
// Each wave: 2 N-tiles x 2 M-subtiles x {pos,neg} -> 8 accumulators (64 VGPRs),
// each B fragment feeds 4 WMMAs. Double-buffered LDS staging (1 barrier/K-tile)
// + global_prefetch_b8 on the adj HBM stream.
// ---------------------------------------------------------------------------
#define PFD (8 * 32)   // prefetch distance in adj columns (8 K-tiles)

__global__ void k_attn_gemm(const float* __restrict__ adj, const float* __restrict__ s1g,
                            const float* __restrict__ s2g, const float* __restrict__ mpg,
                            const float* __restrict__ ilpg, const float* __restrict__ mng,
                            const float* __restrict__ ilng,
                            const unsigned short* __restrict__ inpT,
                            unsigned short* __restrict__ hd) {
  __shared__ __align__(32) unsigned short Pp[2][32 * 32];
  __shared__ __align__(32) unsigned short Pn[2][32 * 32];
  const int tid  = threadIdx.x;
  const int m0   = blockIdx.x * 32;
  const int lane = tid & 31;
  const int wv   = tid >> 5;          // 0..15

  // staging assignment: thread handles rows r0 and r0+16 at column col
  const int r0  = tid >> 5;           // 0..15
  const int col = tid & 31;           // 0..31
  float s1r[2], mpr[2], ilpr[2], mnr[2], ilnr[2];
  long  rofs[2];
  int   ridx[2];
#pragma unroll
  for (int p = 0; p < 2; ++p) {
    const int r = r0 + p * 16;
    ridx[p] = r;
    s1r[p]  = s1g[m0 + r];
    mpr[p]  = mpg[m0 + r];
    ilpr[p] = ilpg[m0 + r];
    mnr[p]  = mng[m0 + r];
    ilnr[p] = ilng[m0 + r];
    rofs[p] = (long)(m0 + r) * NN;
  }

  // WMMA fragment lane mapping
  const int am = lane & 15;     // row (A) / column (B) within tile
  const int kh = lane >> 4;     // K-half selector

  v8f zero = {0.f, 0.f, 0.f, 0.f, 0.f, 0.f, 0.f, 0.f};
  v8f accp[2][2], accn[2][2];
#pragma unroll
  for (int mt = 0; mt < 2; ++mt)
#pragma unroll
    for (int i = 0; i < 2; ++i) { accp[mt][i] = zero; accn[mt][i] = zero; }

  for (int kt = 0; kt < NN / 32; ++kt) {
    const int k0  = kt * 32;
    const int buf = kt & 1;

    // ---- stage probability tiles (bf16) into LDS ----
    {
      const float s2v = s2g[k0 + col];
#pragma unroll
      for (int p = 0; p < 2; ++p) {
        float x = s1r[p] + s2v;
        float e = x > 0.f ? x : GAT_ALPHA * x;
        float aj = adj[rofs[p] + k0 + col];
        float fp = aj > 0.f ? e  : NEGINF;
        float fn = aj > 0.f ? -e : NEGINF;
        Pp[buf][ridx[p] * 32 + col] = f2bf(__expf(fp - mpr[p]) * ilpr[p]);
        Pn[buf][ridx[p] * 32 + col] = f2bf(-__expf(fn - mnr[p]) * ilnr[p]);
      }
    }
    // ---- software prefetch of adj stream (global_prefetch_b8) ----
    if (k0 + PFD < NN) {
#pragma unroll
      for (int p = 0; p < 2; ++p)
        __builtin_prefetch(adj + rofs[p] + k0 + PFD + col, 0, 1);
    }
    __syncthreads();

    // ---- A fragments for both 16-row sub-tiles, pos and neg ----
    Frag Ap[2], An[2];
#pragma unroll
    for (int mt = 0; mt < 2; ++mt) {
      const int rb = (mt * 16 + am) * 32;
      Ap[mt].h[0] = *(const v8u*)&Pp[buf][rb + kh * 8];
      Ap[mt].h[1] = *(const v8u*)&Pp[buf][rb + 16 + kh * 8];
      An[mt].h[0] = *(const v8u*)&Pn[buf][rb + kh * 8];
      An[mt].h[1] = *(const v8u*)&Pn[buf][rb + 16 + kh * 8];
    }

#pragma unroll
    for (int nt = 0; nt < 2; ++nt) {
      const int ncol = (wv * 2 + nt) * 16 + am;
      const unsigned short* bp = inpT + (long)ncol * NN + k0 + kh * 8;
      Frag B;
      B.h[0] = *(const v8u*)bp;
      B.h[1] = *(const v8u*)(bp + 16);
#pragma unroll
      for (int mt = 0; mt < 2; ++mt) {
        accp[mt][nt] = __builtin_amdgcn_wmma_f32_16x16x32_bf16(
            false, Ap[mt].v, false, B.v, (short)0, accp[mt][nt], false, false);
        accn[mt][nt] = __builtin_amdgcn_wmma_f32_16x16x32_bf16(
            false, An[mt].v, false, B.v, (short)0, accn[mt][nt], false, false);
      }
    }
  }

  // store concat [pos | neg] as bf16 rows of width 1024
#pragma unroll
  for (int mt = 0; mt < 2; ++mt) {
#pragma unroll
    for (int nt = 0; nt < 2; ++nt) {
      const int cb = (wv * 2 + nt) * 16 + am;
#pragma unroll
      for (int r = 0; r < 8; ++r) {
        const int row = m0 + mt * 16 + kh * 8 + r;
        hd[(long)row * 1024 + cb]       = f2bf(accp[mt][nt][r]);
        hd[(long)row * 1024 + 512 + cb] = f2bf(accn[mt][nt][r]);
      }
    }
  }
}

// ---------------------------------------------------------------------------
// Kernel 6: new_h = ELU(hd @ wtrans)   M=4096, K=1024, N=512 (bf16 WMMA)
// ---------------------------------------------------------------------------
__global__ void k_out_gemm(const unsigned short* __restrict__ hd,
                           const unsigned short* __restrict__ wT,
                           float* __restrict__ out) {
  const int tid  = threadIdx.x;
  const int lane = tid & 31;
  const int wv   = tid >> 5;
  const int tile = blockIdx.x * 8 + wv;     // 8192 tiles
  const int m0   = (tile >> 5) * 16;
  const int n0   = (tile & 31) * 16;
  const int am   = lane & 15;
  const int kh   = lane >> 4;

  v8f acc = {0.f, 0.f, 0.f, 0.f, 0.f, 0.f, 0.f, 0.f};
  for (int kt = 0; kt < 1024 / 32; ++kt) {
    const int k0 = kt * 32;
    Frag A, B;
    const unsigned short* ap = hd + (long)(m0 + am) * 1024 + k0;
    A.h[0] = *(const v8u*)(ap + kh * 8);
    A.h[1] = *(const v8u*)(ap + 16 + kh * 8);
    const unsigned short* bp = wT + (long)(n0 + am) * 1024 + k0 + kh * 8;
    B.h[0] = *(const v8u*)bp;
    B.h[1] = *(const v8u*)(bp + 16);
    acc = __builtin_amdgcn_wmma_f32_16x16x32_bf16(
        false, A.v, false, B.v, (short)0, acc, false, false);
  }
#pragma unroll
  for (int r = 0; r < 8; ++r) {
    const int row = m0 + kh * 8 + r;
    float v = acc[r];
    out[(long)row * FF + n0 + am] = v > 0.f ? v : (__expf(v) - 1.f);
  }
}

// ---------------------------------------------------------------------------
extern "C" void kernel_launch(void* const* d_in, const int* in_sizes, int n_in,
                              void* d_out, int out_size, void* d_ws, size_t ws_size,
                              hipStream_t stream) {
  (void)in_sizes; (void)n_in; (void)out_size; (void)ws_size;
  const float* inp = (const float*)d_in[0];
  const float* adj = (const float*)d_in[1];
  const float* W   = (const float*)d_in[2];
  const float* a   = (const float*)d_in[3];
  const float* wtr = (const float*)d_in[4];
  float* out = (float*)d_out;

  char* ws = (char*)d_ws;
  float* Wa1 = (float*)(ws + 0);
  float* Wa2 = (float*)(ws + 2048);
  float* s1  = (float*)(ws + 4096);
  float* s2  = (float*)(ws + 20480);
  float* mp  = (float*)(ws + 36864);
  float* ilp = (float*)(ws + 53248);
  float* mn  = (float*)(ws + 69632);
  float* iln = (float*)(ws + 86016);
  unsigned short* inpT = (unsigned short*)(ws + 131072);                      // 4 MB
  unsigned short* wT   = (unsigned short*)(ws + 131072 + 4194304);            // 1 MB
  unsigned short* hd   = (unsigned short*)(ws + 131072 + 4194304 + 1048576);  // 8 MB

  k_wa<<<2, 256, 0, stream>>>(W, a, Wa1, Wa2);
  k_scores<<<512, 256, 0, stream>>>(inp, Wa1, Wa2, s1, s2);
  k_cvt_inpT<<<(512 * 4096) / 256, 256, 0, stream>>>(inp, inpT);
  k_cvt_wT<<<(512 * 1024) / 256, 256, 0, stream>>>(wtr, wT);
  k_stats<<<512, 256, 0, stream>>>(adj, s1, s2, mp, ilp, mn, iln);
  k_attn_gemm<<<128, 512, 0, stream>>>(adj, s1, s2, mp, ilp, mn, iln, inpT, hd);
  k_out_gemm<<<1024, 256, 0, stream>>>(hd, wT, out);
}
